// DecoderLayer_37014028157210
// MI455X (gfx1250) — matrix operand order repaired and checked by reading, not compiled, not run on hardware
//
// CDNA5 / gfx1250 decoder layer: bf16 WMMA GEMMs (v_wmma_f32_16x16x32_bf16),
// VALU online-softmax for the tiny attention cores, wave32 LN reductions.
// All work launched on `stream`; d_ws scratch ~925 MiB; graph-capture safe.
#include <hip/hip_runtime.h>
#include <math.h>

#define NHh 4
#define HDh 32

typedef __attribute__((ext_vector_type(16))) __bf16 v16bf;
typedef __attribute__((ext_vector_type(8)))  float  v8f;

__device__ __forceinline__ unsigned short f2bf(float f) {
  unsigned int u = __float_as_uint(f);
  u += 0x7FFFu + ((u >> 16) & 1u);            // round-to-nearest-even
  return (unsigned short)(u >> 16);
}
__device__ __forceinline__ float gelu_f(float x) {
  return 0.5f * x * (1.0f + erff(x * 0.7071067811865475f));
}

// ---------------------------------------------------------------------------
// Generic batched GEMM: for group g, C_g[M,N] = A_g[M,K](bf16) x W_widx[N,K]^T
// + bias.  A base = A + (aMod ? g%aMod : g)*aStride ; widx = wMod>1 ? g%wMod:0.
// Block = 256 threads = 8 waves; each wave owns one 16x16 tile via WMMA.
// mode: 0 = store acc+bias ; 1 = store gelu(acc+bias).
// ---------------------------------------------------------------------------
__global__ void gemm_bf16_wmma(const __bf16* __restrict__ A, int aMod, long long aStride, int M,
                               const __bf16* __restrict__ Wt, int wMod, long long wStride,
                               const float* __restrict__ bias, long long biasStride,
                               int K, int N,
                               float* __restrict__ Cf, unsigned short* __restrict__ Cb,
                               long long cStride, int mode) {
  const int g    = blockIdx.z;
  const int lane = threadIdx.x & 31;
  const int wave = threadIdx.x >> 5;
  const int n0   = (blockIdx.x * 8 + wave) * 16;
  if (n0 >= N) return;                                   // wave-uniform: EXEC stays full
  const int row0 = blockIdx.y * 16;

  const __bf16* Ab = A + (long long)(aMod ? (g % aMod) : g) * aStride;
  const int widx   = (wMod > 1) ? (g % wMod) : 0;
  const __bf16* Wb = Wt + (long long)widx * wStride;

  int arow = row0 + (lane & 15); if (arow >= M) arow = M - 1;   // clamp pad rows
  int brow = n0   + (lane & 15); if (brow >= N) brow = N - 1;
  const int koffA = (lane >> 4) * 8;    // A 16x32 bf16 layout (ISA 7.12.2)
  const int koffB = (lane >> 4) * 8;    // B 32x16 bf16 layout, same k order as A

  v8f acc = {};
  for (int k0 = 0; k0 < K; k0 += 32) {
    const __bf16* pa = Ab + (long long)arow * K + (k0 + koffA);
    const __bf16* pb = Wb + (long long)brow * K + (k0 + koffB);
    v16bf fa, fb;
#pragma unroll
    for (int i = 0; i < 8; ++i) { fa[i] = pa[i]; fa[i + 8] = pa[i + 16]; }
#pragma unroll
    for (int i = 0; i < 8; ++i) { fb[i] = pb[i]; fb[i + 8] = pb[i + 16]; }
    acc = __builtin_amdgcn_wmma_f32_16x16x32_bf16(false, fa, false, fb,
                                                  (short)0, acc, false, false);
  }

  const int col = n0 + (lane & 15);
  if (col < N) {
    const float bv = bias ? bias[(long long)widx * biasStride + col] : 0.0f;
    const long long cbase = (long long)g * cStride;
#pragma unroll
    for (int r = 0; r < 8; ++r) {
      const int row = row0 + ((lane >> 4) << 3) + r;    // C/D f32 layout
      if (row < M) {
        float v = acc[r] + bv;
        if (mode == 1) v = gelu_f(v);
        const long long ci = cbase + (long long)row * N + col;
        if (Cf) Cf[ci] = v;
        if (Cb) Cb[ci] = f2bf(v);
      }
    }
  }
}

// ---------------------------------------------------------------------------
// Streaming (online-softmax) attention. Q:[G,Lq,128] K/V:[G,Lk,128], head
// h covers cols h*32..h*32+31.  One thread per (g,h,q).  Output -> bf16.
// ---------------------------------------------------------------------------
__global__ void attn_online(const float* __restrict__ Q, const float* __restrict__ Km,
                            const float* __restrict__ Vm, unsigned short* __restrict__ Ob,
                            int G, int Lq, int Lk, float scale) {
  long long t = (long long)blockIdx.x * blockDim.x + threadIdx.x;
  int qi = (int)(t % Lq); long long r = t / Lq;
  int h  = (int)(r % NHh); long long g = r / NHh;
  if (g >= G) return;
  const float* qp = Q + ((long long)g * Lq + qi) * 128 + h * HDh;
  float qv[HDh];
#pragma unroll
  for (int e = 0; e < HDh; ++e) qv[e] = qp[e];
  const float* kp = Km + (long long)g * Lk * 128 + h * HDh;
  const float* vp = Vm + (long long)g * Lk * 128 + h * HDh;
  float m = -3.0e38f, s = 0.0f, o[HDh];
#pragma unroll
  for (int e = 0; e < HDh; ++e) o[e] = 0.0f;
  for (int k = 0; k < Lk; ++k) {
    const float* kr = kp + (long long)k * 128;
    float dot = 0.0f;
#pragma unroll
    for (int e = 0; e < HDh; ++e) dot += qv[e] * kr[e];
    dot *= scale;
    float mn = fmaxf(m, dot);
    float c  = __expf(m - mn);
    float p  = __expf(dot - mn);
    s = s * c + p;
    const float* vr = vp + (long long)k * 128;
#pragma unroll
    for (int e = 0; e < HDh; ++e) o[e] = o[e] * c + p * vr[e];
    m = mn;
  }
  float inv = 1.0f / s;
  unsigned short* op = Ob + ((long long)g * Lq + qi) * 128 + h * HDh;
#pragma unroll
  for (int e = 0; e < HDh; ++e) op[e] = f2bf(o[e] * inv);
}

// ---------------------------------------------------------------------------
// y = LN(alpha*X + Y) * gamma + beta over rows of 128. One wave32 per row.
// gamma row index = (row/gDiv)%gMod (per-D params) or 0 (shared, gMod<=1).
// ---------------------------------------------------------------------------
__global__ void resid_ln(const float* __restrict__ X, const float* __restrict__ Y, float alpha,
                         const float* __restrict__ gamma, const float* __restrict__ beta,
                         int gDiv, int gMod,
                         float* __restrict__ Of, unsigned short* __restrict__ Ob,
                         long long nRows) {
  long long row = (long long)blockIdx.x * (blockDim.x >> 5) + (threadIdx.x >> 5);
  int lane = threadIdx.x & 31;
  if (row >= nRows) return;
  const long long base = row * 128;
  float v[4]; float s = 0.f;
#pragma unroll
  for (int i = 0; i < 4; ++i) {
    int c = lane + 32 * i;
    float a = alpha * X[base + c] + (Y ? Y[base + c] : 0.0f);
    v[i] = a; s += a;
  }
#pragma unroll
  for (int o = 16; o > 0; o >>= 1) s += __shfl_xor(s, o, 32);
  float mean = s * (1.0f / 128.0f);
  float vs = 0.f;
#pragma unroll
  for (int i = 0; i < 4; ++i) { float d = v[i] - mean; vs += d * d; }
#pragma unroll
  for (int o = 16; o > 0; o >>= 1) vs += __shfl_xor(vs, o, 32);
  float rstd = rsqrtf(vs * (1.0f / 128.0f) + 1e-5f);
  long long gi = (gMod > 1) ? ((row / gDiv) % gMod) : 0;
#pragma unroll
  for (int i = 0; i < 4; ++i) {
    int c = lane + 32 * i;
    float out = (v[i] - mean) * rstd * gamma[gi * 128 + c] + beta[gi * 128 + c];
    if (Of) Of[base + c] = out;
    if (Ob) Ob[base + c] = f2bf(out);
  }
}

// (B,D,S,d) <-> (B,S,D,d) transpose, optional simultaneous bf16 copy.
__global__ void transpose_tok(const float* __restrict__ in, float* __restrict__ Of,
                              unsigned short* __restrict__ Ob, int dir, long long total) {
  long long idx = (long long)blockIdx.x * blockDim.x + threadIdx.x;
  if (idx >= total) return;
  int c = (int)(idx & 127);
  long long t = idx >> 7, it;
  if (dir == 0) { long long Dj = t % 768, r = t / 768, s = r % 10, b = r / 10;
                  it = (b * 768 + Dj) * 10 + s; }
  else          { long long s = t % 10, r = t / 10, Dj = r % 768, b = r / 768;
                  it = (b * 10 + s) * 768 + Dj; }
  float v = in[it * 128 + c];
  if (Of) Of[idx] = v;
  if (Ob) Ob[idx] = f2bf(v);
}

__global__ void cvt_f32_bf16(const float* __restrict__ in, unsigned short* __restrict__ out,
                             long long n) {
  long long i = (long long)blockIdx.x * blockDim.x + threadIdx.x;
  if (i < n) out[i] = f2bf(in[i]);
}

// in f32 [G,K,N] -> out bf16 [G,N,K]  (weights into WMMA B-friendly layout)
__global__ void wt_transpose(const float* __restrict__ in, unsigned short* __restrict__ out,
                             int Kd, int Nd, long long total) {
  long long idx = (long long)blockIdx.x * blockDim.x + threadIdx.x;
  if (idx >= total) return;
  long long k = idx % Kd, r = idx / Kd, n = r % Nd, g = r / Nd;
  out[idx] = f2bf(in[(g * Kd + k) * Nd + n]);
}

// ---------------------------------------------------------------------------
static void g_gemm(hipStream_t st, const void* A, int aMod, long long aStr, int M,
                   const void* Wt, int wMod, long long wStr,
                   const float* bias, long long bStr, int K, int N,
                   float* Cf, unsigned short* Cb, long long cStr, int mode, int G) {
  dim3 grid((unsigned)(((N + 15) / 16 + 7) / 8), (unsigned)((M + 15) / 16), (unsigned)G);
  gemm_bf16_wmma<<<grid, dim3(256), 0, st>>>((const __bf16*)A, aMod, aStr, M,
                                             (const __bf16*)Wt, wMod, wStr,
                                             bias, bStr, K, N, Cf, Cb, cStr, mode);
}

extern "C" void kernel_launch(void* const* d_in, const int* in_sizes, int n_in,
                              void* d_out, int out_size, void* d_ws, size_t ws_size,
                              hipStream_t stream) {
  (void)in_sizes; (void)n_in; (void)out_size; (void)ws_size;
  const float* x     = (const float*)d_in[0];
  const float* cross = (const float*)d_in[1];
  const float* timeW = (const float*)d_in[2];
  const float* timeB = (const float*)d_in[3];
  const float* sendW = (const float*)d_in[4];
  const float* sendB = (const float*)d_in[5];
  const float* recvW = (const float*)d_in[6];
  const float* recvB = (const float*)d_in[7];
  const float* router= (const float*)d_in[8];
  const float* lnG   = (const float*)d_in[9];
  const float* lnB   = (const float*)d_in[10];
  const float* mlpW1 = (const float*)d_in[11];
  const float* mlpB1 = (const float*)d_in[12];
  const float* mlpW2 = (const float*)d_in[13];
  const float* mlpB2 = (const float*)d_in[14];
  const float* caW   = (const float*)d_in[15];
  const float* caB   = (const float*)d_in[16];
  const float* ln1G  = (const float*)d_in[17];
  const float* ln1B  = (const float*)d_in[18];
  const float* ln31G = (const float*)d_in[19];
  const float* ln31B = (const float*)d_in[20];
  const float* m2W1  = (const float*)d_in[21];
  const float* m2B1  = (const float*)d_in[22];
  const float* m2W2  = (const float*)d_in[23];
  const float* m2B2  = (const float*)d_in[24];
  const float* predW = (const float*)d_in[25];
  const float* predB = (const float*)d_in[26];

  const long long NTOK = 32LL * 768 * 10;           // 245760 tokens, d=128
  const int GF = (int)(NTOK / 16);                  // flat-GEMM groups (M=16)
  const int GQ = 32 * 10;                           // dim-stage groups (b,s)
  const int GC = 32 * 768;                          // cross-stage groups (b,D)

  // ---- scratch layout ----
  char* ws = (char*)d_ws; size_t off = 0;
  auto alloc = [&](size_t bytes) -> void* {
    off = (off + 255) & ~(size_t)255; void* p = ws + off; off += bytes; return p;
  };
  float* bufX = (float*)alloc((size_t)NTOK * 128 * 4);
  float* bufQ = (float*)alloc((size_t)NTOK * 128 * 4);
  float* bufK = (float*)alloc((size_t)NTOK * 128 * 4);
  float* bufV = (float*)alloc((size_t)NTOK * 128 * 4);
  unsigned short* abf = (unsigned short*)alloc((size_t)NTOK * 128 * 2);
  unsigned short* cbf = (unsigned short*)alloc((size_t)NTOK * 128 * 2);
  unsigned short* hbf = (unsigned short*)alloc((size_t)NTOK * 256 * 2);
  unsigned short* timeWt = (unsigned short*)alloc((size_t)4 * 128 * 128 * 2);
  unsigned short* sendWt = (unsigned short*)alloc((size_t)4 * 128 * 128 * 2);
  unsigned short* recvWt = (unsigned short*)alloc((size_t)4 * 128 * 128 * 2);
  unsigned short* mlp1t  = (unsigned short*)alloc((size_t)2 * 256 * 128 * 2);
  unsigned short* mlp2t  = (unsigned short*)alloc((size_t)2 * 128 * 256 * 2);
  unsigned short* caWt   = (unsigned short*)alloc((size_t)768 * 4 * 128 * 128 * 2);
  unsigned short* m2W1t  = (unsigned short*)alloc((size_t)768 * 256 * 128 * 2);
  unsigned short* m2W2t  = (unsigned short*)alloc((size_t)768 * 128 * 256 * 2);
  unsigned short* predWt = (unsigned short*)alloc((size_t)768 * 24 * 128 * 2);
  unsigned short* rbf    = (unsigned short*)alloc((size_t)10 * 10 * 128 * 2);
  float* S1 = (float*)alloc((size_t)320 * 10 * 128 * 4);
  float* S2 = (float*)alloc((size_t)320 * 10 * 128 * 4);
  float* S3 = (float*)alloc((size_t)320 * 10 * 128 * 4);
  float* S4 = (float*)alloc((size_t)320 * 10 * 128 * 4);
  unsigned short* S1b = (unsigned short*)alloc((size_t)320 * 10 * 128 * 2);
  unsigned short* S2b = (unsigned short*)alloc((size_t)320 * 10 * 128 * 2);

  auto cvt = [&](const float* in, unsigned short* out, long long n) {
    cvt_f32_bf16<<<dim3((unsigned)((n + 255) / 256)), dim3(256), 0, stream>>>(in, out, n);
  };
  auto wtr = [&](const float* in, unsigned short* out, int Kd, int Nd, long long Gn) {
    long long tot = Gn * Kd * Nd;
    wt_transpose<<<dim3((unsigned)((tot + 255) / 256)), dim3(256), 0, stream>>>(in, out, Kd, Nd, tot);
  };
  auto attn = [&](const float* Qm, const float* Km, const float* Vm, unsigned short* Ob,
                  int G, int Lq, int Lk) {
    long long tot = (long long)G * NHh * Lq;
    attn_online<<<dim3((unsigned)((tot + 255) / 256)), dim3(256), 0, stream>>>(
        Qm, Km, Vm, Ob, G, Lq, Lk, 0.17677669529663687f);
  };
  auto ln = [&](const float* X, const float* Y, float alpha, const float* g_, const float* b_,
                int gDiv, int gMod, float* Of, unsigned short* Ob) {
    resid_ln<<<dim3((unsigned)((NTOK + 7) / 8)), dim3(256), 0, stream>>>(
        X, Y, alpha, g_, b_, gDiv, gMod, Of, Ob, NTOK);
  };
  auto tr = [&](const float* in, float* Of, unsigned short* Ob, int dir) {
    long long tot = NTOK * 128;
    transpose_tok<<<dim3((unsigned)((tot + 255) / 256)), dim3(256), 0, stream>>>(in, Of, Ob, dir, tot);
  };

  // ---- weight prep (f32 -> bf16 [N,K]) ----
  wtr(timeW, timeWt, 128, 128, 4);
  wtr(sendW, sendWt, 128, 128, 4);
  wtr(recvW, recvWt, 128, 128, 4);
  wtr(mlpW1, mlp1t, 128, 256, 2);
  wtr(mlpW2, mlp2t, 256, 128, 2);
  wtr(caW,  caWt,  128, 128, 768 * 4);
  wtr(m2W1, m2W1t, 128, 256, 768);
  wtr(m2W2, m2W2t, 256, 128, 768);
  wtr(predW, predWt, 128, 24, 768);
  cvt(router, rbf, 10LL * 10 * 128);
  cvt(x, abf, NTOK * 128);
  cvt(cross, cbf, NTOK * 128);

  // ---- stage 1: time self-attention over S=10 ----
  g_gemm(stream, abf, 0, 16*128, 16, timeWt + 0*16384, 1, 0, timeB + 0,   0, 128, 128, bufQ, 0, 16*128, 0, GF);
  g_gemm(stream, abf, 0, 16*128, 16, timeWt + 1*16384, 1, 0, timeB + 128, 0, 128, 128, bufK, 0, 16*128, 0, GF);
  g_gemm(stream, abf, 0, 16*128, 16, timeWt + 2*16384, 1, 0, timeB + 256, 0, 128, 128, bufV, 0, 16*128, 0, GF);
  attn(bufQ, bufK, bufV, abf, 32 * 768, 10, 10);
  g_gemm(stream, abf, 0, 16*128, 16, timeWt + 3*16384, 1, 0, timeB + 384, 0, 128, 128, bufK, 0, 16*128, 0, GF);
  ln(x, bufK, 1.0f, lnG + 0, lnB + 0, 1, 1, bufX, abf);
  g_gemm(stream, abf, 0, 16*128, 16, mlp1t + 0,     1, 0, mlpB1 + 0,   0, 128, 256, 0, hbf, 16*256, 1, GF);
  g_gemm(stream, hbf, 0, 16*256, 16, mlp2t + 0,     1, 0, mlpB2 + 0,   0, 256, 128, bufK, 0, 16*128, 0, GF);
  ln(bufX, bufK, 1.0f, lnG + 128, lnB + 128, 1, 1, bufX, nullptr);

  // ---- stage 2: dim-stage (router send/recv) over D=768 ----
  tr(bufX, bufQ, abf, 0);                                   // xd (B,S,D,d) f32 + bf16
  g_gemm(stream, rbf, 10, 10*128, 10, sendWt + 0,      1, 0, sendB + 0,   0, 128, 128, S1, 0, 10*128, 0, GQ);
  g_gemm(stream, abf, 0, 16*128, 16, sendWt + 1*16384, 1, 0, sendB + 128, 0, 128, 128, bufK, 0, 16*128, 0, GF);
  g_gemm(stream, abf, 0, 16*128, 16, sendWt + 2*16384, 1, 0, sendB + 256, 0, 128, 128, bufV, 0, 16*128, 0, GF);
  attn(S1, bufK, bufV, S1b, GQ, 10, 768);
  g_gemm(stream, S1b, 0, 10*128, 10, sendWt + 3*16384, 1, 0, sendB + 384, 0, 128, 128, S2, S2b, 10*128, 0, GQ);
  g_gemm(stream, abf, 0, 16*128, 16, recvWt + 0,       1, 0, recvB + 0,   0, 128, 128, bufK, 0, 16*128, 0, GF);
  g_gemm(stream, S2b, 0, 10*128, 10, recvWt + 1*16384, 1, 0, recvB + 128, 0, 128, 128, S3, 0, 10*128, 0, GQ);
  g_gemm(stream, S2b, 0, 10*128, 10, recvWt + 2*16384, 1, 0, recvB + 256, 0, 128, 128, S4, 0, 10*128, 0, GQ);
  attn(bufK, S3, S4, abf, GQ, 768, 10);
  g_gemm(stream, abf, 0, 16*128, 16, recvWt + 3*16384, 1, 0, recvB + 384, 0, 128, 128, bufV, 0, 16*128, 0, GF);
  ln(bufQ, bufV, 1.0f, lnG + 2*128, lnB + 2*128, 1, 1, bufQ, abf);
  g_gemm(stream, abf, 0, 16*128, 16, mlp1t + 32768, 1, 0, mlpB1 + 256, 0, 128, 256, 0, hbf, 16*256, 1, GF);
  g_gemm(stream, hbf, 0, 16*256, 16, mlp2t + 32768, 1, 0, mlpB2 + 128, 0, 256, 128, bufV, 0, 16*128, 0, GF);
  ln(bufQ, bufV, 1.0f, lnG + 3*128, lnB + 3*128, 1, 1, bufQ, nullptr);
  tr(bufQ, bufX, abf, 1);                                   // xs (B,D,S,d) f32 + bf16

  // ---- stage 3: per-D cross-attention + MLP2 + heads ----
  g_gemm(stream, abf, 0, 10*128, 10, caWt + 0*16384, 768, 4*16384, caB + 0,   512, 128, 128, bufQ, 0, 10*128, 0, GC);
  g_gemm(stream, cbf, 0, 10*128, 10, caWt + 1*16384, 768, 4*16384, caB + 128, 512, 128, 128, bufK, 0, 10*128, 0, GC);
  g_gemm(stream, cbf, 0, 10*128, 10, caWt + 2*16384, 768, 4*16384, caB + 256, 512, 128, 128, bufV, 0, 10*128, 0, GC);
  attn(bufQ, bufK, bufV, abf, GC, 10, 10);
  g_gemm(stream, abf, 0, 10*128, 10, caWt + 3*16384, 768, 4*16384, caB + 384, 512, 128, 128, bufK, 0, 10*128, 0, GC);
  ln(bufX, bufK, 1.0f, ln1G, ln1B, 10, 768, bufX, abf);
  g_gemm(stream, abf, 0, 10*128, 10, m2W1t, 768, 256*128, m2B1, 256, 128, 256, 0, hbf, 10*256, 1, GC);
  g_gemm(stream, hbf, 0, 10*256, 10, m2W2t, 768, 128*256, m2B2, 128, 256, 128, bufK, 0, 10*128, 0, GC);

  float* dec  = (float*)d_out;
  float* pred = dec + NTOK * 128;
  ln(bufK, nullptr, 2.0f, ln31G, ln31B, 10, 768, dec, abf); // dec = LN(2*mo)
  g_gemm(stream, abf, 0, 10*128, 10, predWt, 768, 24*128, predB, 24, 128, 24, pred, 0, 10*24, 0, GC);
}
